// sketchDiscriminator_42314017800368
// MI455X (gfx1250) — compile-verified, matrix operand order checked by reading, not run on hardware
//
#include <hip/hip_runtime.h>

// Problem constants (from reference)
#define IN_DIM 5
#define HID 512
#define SEQ 250
#define BATCH 256
#define G4 2048                 // 4*HID
#define KAUG 544                // 512 (h) + 5 (x) + 1 (bias) + pad -> 17*32
#define KSTEPS 17               // KAUG / 32
#define LPAD 552                // LDS row stride in f16 elems: 1104B, 16B-aligned, bank-spread

typedef __attribute__((ext_vector_type(16))) _Float16 v16h;
typedef __attribute__((ext_vector_type(8)))  _Float16 v8h;
typedef __attribute__((ext_vector_type(8)))  float    v8f;

#if defined(__has_builtin) && __has_builtin(__builtin_amdgcn_sched_barrier)
#define SCHED_FENCE() __builtin_amdgcn_sched_barrier(0)
#else
#define SCHED_FENCE()
#endif

__device__ __forceinline__ float sigf(float x)      { return 1.0f / (1.0f + __expf(-x)); }
__device__ __forceinline__ float tanh_fast(float x) { return 1.0f - 2.0f / (__expf(2.0f * x) + 1.0f); }

// ---------------------------------------------------------------------------
// Kernel 1: build augmented f16 weight  Waug[2048][544]:
//   cols [0,512)    = W_hh[g][k]
//   cols [512,517)  = W_ih[g][i]
//   col  517        = b_ih[g] + b_hh[g]
//   cols [518,544)  = 0
// so that  gates = [h | x | 1 | 0] @ Waug^T  in one WMMA sweep.
// ---------------------------------------------------------------------------
__global__ void build_waug(const float* __restrict__ Whh,
                           const float* __restrict__ Wih,
                           const float* __restrict__ bih,
                           const float* __restrict__ bhh,
                           _Float16* __restrict__ Waug)
{
    int idx = blockIdx.x * blockDim.x + threadIdx.x;
    if (idx >= G4 * KAUG) return;
    int col = idx / KAUG;
    int k   = idx % KAUG;
    float v;
    if (k < HID)                 v = Whh[col * HID + k];
    else if (k < HID + IN_DIM)   v = Wih[col * IN_DIM + (k - HID)];
    else if (k == HID + IN_DIM)  v = bih[col] + bhh[col];
    else                         v = 0.0f;
    Waug[idx] = (_Float16)v;
}

// ---------------------------------------------------------------------------
// Kernel 2: persistent LSTM. grid = 16 blocks (one per 16-row batch tile),
// block = 512 threads = 16 wave32 waves. Wave w owns hidden units
// [32w, 32w+32) of all four gates -> 8 WMMA accumulator tiles per wave.
// h lives in LDS (f16); c lives in 16 VGPRs per lane for all 250 steps.
// B tiles stream from L2 each step; address is laundered through a VGPR so
// the compiler cannot hoist the loop-invariant loads (spill hazard), and
// sched_barriers force all 16 b128 loads of a k-step into flight before the
// 8-WMMA burst (one L2 latency amortized over 8 matrix ops).
// ---------------------------------------------------------------------------
__global__ __launch_bounds__(512, 1) void lstm_persist(
    const float* __restrict__ strokes,   // [BATCH][SEQ][IN_DIM] fp32
    const _Float16* __restrict__ Waug,   // [G4][KAUG] f16
    const float* __restrict__ Wout,      // [HID]
    const float* __restrict__ bout,      // [1]
    float* __restrict__ out)             // [BATCH]
{
    __shared__ __align__(16) _Float16 hbuf[16 * LPAD];

    const int tid  = threadIdx.x;
    const int w    = tid >> 5;           // wave id 0..15
    const int lane = tid & 31;
    const int nl   = lane & 15;          // row (A) / column (B) within tile
    const int hi   = lane >> 4;          // half-wave select
    const int koff = hi * 8;             // K offset per ISA 16-bit operand layout
    const int m0   = blockIdx.x * 16;    // batch row base

    // zero LDS (covers h region and zero-pad K tail)
    for (int i = tid; i < 16 * LPAD; i += 512) hbuf[i] = (_Float16)0.0f;
    __syncthreads();
    if (tid < 16) hbuf[tid * LPAD + HID + IN_DIM] = (_Float16)1.0f;  // bias lane "1"

    // operand base pointers (ISA layout: lanes 0-15 K{0..7,16..23}, lanes 16-31 K{8..15,24..31})
    const _Float16* abase = &hbuf[nl * LPAD + koff];
    const _Float16* bbase = Waug + (size_t)(w * 32 + nl) * KAUG + koff;

    float cst[16];
    #pragma unroll
    for (int i = 0; i < 16; ++i) cst[i] = 0.0f;

    for (int t = 0; t < SEQ; ++t) {
        // stage x_t into augmented-K slots [512..517)
        if (tid < 16 * IN_DIM) {
            int m = tid / IN_DIM, i = tid % IN_DIM;
            hbuf[m * LPAD + HID + i] =
                (_Float16)strokes[((size_t)(m0 + m) * SEQ + t) * IN_DIM + i];
        }
        __syncthreads();   // h(t) + x(t) visible to all waves

        // Launder a zero offset through a VGPR so the compiler cannot prove
        // the B loads are loop-invariant (prevents hoist-then-spill-to-scratch).
        int tofs = 0;
        asm volatile("" : "+v"(tofs));
        const _Float16* bptr = bbase + tofs;

        v8f acc[8];
        #pragma unroll
        for (int nt = 0; nt < 8; ++nt) acc[nt] = (v8f){0,0,0,0,0,0,0,0};

        #pragma unroll
        for (int ks = 0; ks < KSTEPS; ++ks) {
            v8h alo = *(const v8h*)(abase + ks * 32);
            v8h ahi = *(const v8h*)(abase + ks * 32 + 16);
            v16h a  = __builtin_shufflevector(alo, ahi,
                        0,1,2,3,4,5,6,7,8,9,10,11,12,13,14,15);

            // gather all 8 B tiles for this k-step (16 b128 loads in flight) ...
            v16h b[8];
            #pragma unroll
            for (int nt = 0; nt < 8; ++nt) {
                const int g  = nt >> 1, tp = nt & 1;
                const int co = (g * 512 + tp * 16) * KAUG;   // compile-time elem offset
                v8h blo = *(const v8h*)(bptr + co + ks * 32);
                v8h bhi = *(const v8h*)(bptr + co + ks * 32 + 16);
                b[nt]   = __builtin_shufflevector(blo, bhi,
                            0,1,2,3,4,5,6,7,8,9,10,11,12,13,14,15);
            }
            SCHED_FENCE();   // keep the load clause above the WMMA burst
            #pragma unroll
            for (int nt = 0; nt < 8; ++nt) {
                acc[nt] = __builtin_amdgcn_wmma_f32_16x16x32_f16(
                              false, a, false, b[nt], (short)0, acc[nt], false, false);
            }
            SCHED_FENCE();   // keep next k-step's loads below this burst
        }
        __syncthreads();   // all reads of h(t) complete before overwrite

        // gate fusion: C-tile layout => (m = r + 8*hi, unit j = w*32 + tp*16 + nl)
        #pragma unroll
        for (int tp = 0; tp < 2; ++tp) {
            #pragma unroll
            for (int r = 0; r < 8; ++r) {
                float gi = acc[0 + tp][r];
                float gf = acc[2 + tp][r];
                float gg = acc[4 + tp][r];
                float go = acc[6 + tp][r];
                float c  = sigf(gf) * cst[tp * 8 + r] + sigf(gi) * tanh_fast(gg);
                cst[tp * 8 + r] = c;
                float h  = sigf(go) * tanh_fast(c);
                hbuf[(r + 8 * hi) * LPAD + (w * 32 + tp * 16 + nl)] = (_Float16)h;
            }
        }
    }
    __syncthreads();

    // output head: wave w reduces row w of the tile
    float sum = 0.0f;
    for (int u = lane; u < HID; u += 32)
        sum += (float)hbuf[w * LPAD + u] * Wout[u];
    #pragma unroll
    for (int off = 16; off > 0; off >>= 1)
        sum += __shfl_down(sum, off, 32);
    if (lane == 0) {
        float raw = sum + bout[0];
        float lr  = raw > 0.0f ? raw : 0.1f * raw;
        out[m0 + w] = sigf(lr);
    }
}

// ---------------------------------------------------------------------------
extern "C" void kernel_launch(void* const* d_in, const int* in_sizes, int n_in,
                              void* d_out, int out_size, void* d_ws, size_t ws_size,
                              hipStream_t stream)
{
    const float* strokes = (const float*)d_in[0];
    const float* W_ih    = (const float*)d_in[1];
    const float* W_hh    = (const float*)d_in[2];
    const float* b_ih    = (const float*)d_in[3];
    const float* b_hh    = (const float*)d_in[4];
    const float* W_out   = (const float*)d_in[5];
    const float* b_out   = (const float*)d_in[6];

    _Float16* Waug = (_Float16*)d_ws;    // needs G4*KAUG*2 = ~2.2 MB of workspace

    int total = G4 * KAUG;
    build_waug<<<(total + 255) / 256, 256, 0, stream>>>(W_hh, W_ih, b_ih, b_hh, Waug);
    lstm_persist<<<BATCH / 16, 512, 0, stream>>>(strokes, Waug, W_out, b_out,
                                                 (float*)d_out);
}
